// Attention_50371376447733
// MI455X (gfx1250) — matrix operand order repaired and checked by reading, not compiled, not run on hardware
//
#include <hip/hip_runtime.h>
#include <hip/hip_bf16.h>

typedef __attribute__((ext_vector_type(16))) __bf16   v16bf;
typedef __attribute__((ext_vector_type(8)))  __bf16   v8bf;
typedef __attribute__((ext_vector_type(8)))  float    v8f;
typedef __attribute__((ext_vector_type(4)))  int      v4i;
typedef __attribute__((ext_vector_type(8)))  int      v8i;
typedef __attribute__((ext_vector_type(4)))  unsigned v4u;

#define WMMA_BF16(a, b, c) \
  __builtin_amdgcn_wmma_f32_16x16x32_bf16(false, (a), false, (b), (short)0, (c), false, false)

// ---- gfx1250 data-movement feature detection (all guarded with fallbacks) ----
#ifdef __HIP_DEVICE_COMPILE__
#  if __has_builtin(__builtin_amdgcn_tensor_load_to_lds) && __has_builtin(__builtin_amdgcn_s_wait_tensorcnt)
#    define HAVE_TDM 1
#  else
#    define HAVE_TDM 0
#  endif
#  if __has_builtin(__builtin_amdgcn_global_load_async_to_lds_b128)
#    define HAVE_ASYNC_LDS 1
#  else
#    define HAVE_ASYNC_LDS 0
#  endif
#else
#  define HAVE_TDM 0
#  define HAVE_ASYNC_LDS 0
#endif
#define HAVE_RAWBUF (HAVE_TDM || HAVE_ASYNC_LDS)

namespace {
constexpr int B      = 2;
constexpr int H      = 16;
constexpr int S      = 2048;
constexpr int D      = 64;
constexpr int QTILE  = 128;   // rows per block (8 waves x 16)
constexpr int KTILE  = 32;    // keys per staged tile
constexpr int VST    = 48;    // padded stride for transposed V (keeps 32B alignment)
constexpr float NEGMAX = -3.0e38f;
constexpr float EPS    = 1e-12f;
// cosine-sim attention: |logit| <= 8 (L2-normalized q,k; scale 8; unit learned scales).
// A static shift makes softmax exact without a running max (margin for bf16 rounding).
constexpr float SMAX   = 10.0f;
}

#if HAVE_TDM
typedef __attribute__((address_space(3))) char lds_char;
// TDM: one descriptor-driven DMA moves a 32x64 fp32 tile (global, row-major) into LDS.
// D# packing per CDNA5 ISA sect. 8.3/8.4 (group0 128b, group1 256b).
__device__ __forceinline__ void tdm_copy_tile(const float* gsrc, float* ldst) {
  const unsigned lds_addr = (unsigned)(size_t)(lds_char*)(void*)ldst;
  const unsigned long long ga = (unsigned long long)(size_t)gsrc;
  v4u g0;
  g0.x = 1u;                                              // count=1, user descriptor
  g0.y = lds_addr;                                        // [63:32] lds_addr
  g0.z = (unsigned)ga;                                    // [95:64] global_addr lo
  g0.w = (unsigned)((ga >> 32) & 0x01FFFFFFu) | (2u << 30); // [120:96] ga hi, [127:126] type=2
  v8i g1;
  g1[0] = (int)(2u << 16);        // workgroup_mask=0 (not in cluster), data_size=2 (4 bytes)
  g1[1] = (int)(64u << 16);       // tensor_dim0 = 64 (bits 79:48, low half)
  g1[2] = (int)(KTILE << 16);     // tensor_dim0 hi = 0, tensor_dim1 = 32 (bits 111:80, low half)
  g1[3] = (int)(64u << 16);       // tensor_dim1 hi = 0, tile_dim0 = 64 (bits 127:112)
  g1[4] = (int)KTILE;             // tile_dim1 = 32, tile_dim2 = 0
  g1[5] = 64;                     // tensor_dim0_stride = 64 elements (bits 207:160, low)
  g1[6] = 0;                      // stride0 hi = 0, tensor_dim1_stride lo = 0
  g1[7] = 0;
  const v4i gz4 = {0, 0, 0, 0};               // groups 2/3 unused (<=2D tensor)
  const v8i gz8 = {0, 0, 0, 0, 0, 0, 0, 0};   // extra group (6-arg toolchain form)
  __builtin_amdgcn_tensor_load_to_lds(g0, g1, gz4, gz4, gz8, 0);
}
__device__ __forceinline__ void wait_copies() {
  __builtin_amdgcn_s_wait_tensorcnt((short)0);   // NOP for waves that issued nothing
}
#elif HAVE_ASYNC_LDS
typedef __attribute__((address_space(1))) v4i g_v4i;
typedef __attribute__((address_space(3))) v4i l_v4i;
__device__ __forceinline__ void async_copy32B(const float* g, float* l) {
  g_v4i* gp = (g_v4i*)g;
  l_v4i* lp = (l_v4i*)l;
  __builtin_amdgcn_global_load_async_to_lds_b128(gp, lp, 0, 0);
  __builtin_amdgcn_global_load_async_to_lds_b128(gp, lp, 16, 0);
}
__device__ __forceinline__ void wait_copies() {
#if __has_builtin(__builtin_amdgcn_s_wait_asynccnt)
  __builtin_amdgcn_s_wait_asynccnt(0);
#elif defined(__gfx1250__)
  asm volatile("s_wait_asynccnt 0x0" ::: "memory");
#endif
}
#endif

__global__ __launch_bounds__(256)
void attn_cossim_fwd(const float* __restrict__ q, const float* __restrict__ k,
                     const float* __restrict__ v, const float* __restrict__ qscale,
                     const float* __restrict__ kscale, const unsigned char* __restrict__ mask,
                     float* __restrict__ out)
{
  __shared__ __align__(32) __bf16 Klds[KTILE * D];       // 4 KB  normalized K (row-major bf16)
  __shared__ __align__(32) __bf16 Vtlds[D * VST];        // 6 KB  V transposed [dim][key] bf16
  __shared__ __align__(32) __bf16 Pscr[8 * 16 * KTILE];  // 8 KB  per-wave P scratch
  __shared__ unsigned char Mlds[KTILE];
#if HAVE_RAWBUF
  __shared__ __align__(16) float Kraw[2][KTILE * D];     // 16 KB double-buffered raw K
  __shared__ __align__(16) float Vraw[2][KTILE * D];     // 16 KB double-buffered raw V
#endif

  const int tid  = threadIdx.x;
  const int wave = tid >> 5;
  const int lane = tid & 31;
  const int hl   = lane >> 4;   // lane half
  const int ln   = lane & 15;

  const int bh   = blockIdx.y;
  const int bidx = bh >> 4;               // batch (H == 16)
  const int qb   = blockIdx.x * QTILE;
  const size_t base = (size_t)bh * S * D;

  // ---- Q: load, L2-normalize, apply q_scale + cosine scale 8, bf16 A-fragment layout ----
  const int qrow = qb + wave * 16 + ln;   // A-layout: M = lane&15
  const float* qp = q + base + (size_t)qrow * D;
  const int off0 = hl * 8;
  float r0[8], r1[8], r2[8], r3[8];
  #pragma unroll
  for (int e = 0; e < 8; ++e) {
    r0[e] = qp[off0 + e];
    r1[e] = qp[16 + off0 + e];
    r2[e] = qp[32 + off0 + e];
    r3[e] = qp[48 + off0 + e];
  }
  float ss = 0.f;
  #pragma unroll
  for (int e = 0; e < 8; ++e)
    ss += r0[e]*r0[e] + r1[e]*r1[e] + r2[e]*r2[e] + r3[e]*r3[e];
  ss += __shfl_xor(ss, 16);
  const float qinv = 8.0f / fmaxf(sqrtf(ss), EPS);
  v16bf aQ0, aQ1;
  #pragma unroll
  for (int e = 0; e < 8; ++e) {
    aQ0[e]     = (__bf16)(r0[e] * qinv * qscale[off0 + e]);
    aQ0[e + 8] = (__bf16)(r1[e] * qinv * qscale[16 + off0 + e]);
    aQ1[e]     = (__bf16)(r2[e] * qinv * qscale[32 + off0 + e]);
    aQ1[e + 8] = (__bf16)(r3[e] * qinv * qscale[48 + off0 + e]);
  }

  // all-ones B matrix: row-sum of P computed by the WMMA pipe (no cross-lane reductions)
  v16bf vones;
  #pragma unroll
  for (int e = 0; e < 16; ++e) vones[e] = (__bf16)1.0f;

  // staging coords: 256 threads x (1 key-row slice of 8 dims) = 32x64 tile
  const int sr = tid >> 3;
  const int sc = (tid & 7) * 8;
  float ksc[8];
  #pragma unroll
  for (int e = 0; e < 8; ++e) ksc[e] = kscale[sc + e];

  v8f o0 = {}, o1 = {}, o2 = {}, o3 = {}, osum = {};

  const int nkt = (qb + QTILE) / KTILE;   // causal tile limit for this block

#if HAVE_TDM
  if (wave == 0) {   // wave-level DMA issue (EXEC ignored by TENSOR ops)
    tdm_copy_tile(k + base, &Kraw[0][0]);
    tdm_copy_tile(v + base, &Vraw[0][0]);
  }
#elif HAVE_ASYNC_LDS
  async_copy32B(k + base + (size_t)sr * D + sc, &Kraw[0][sr * D + sc]);
  async_copy32B(v + base + (size_t)sr * D + sc, &Vraw[0][sr * D + sc]);
#endif

  for (int kt = 0; kt < nkt; ++kt) {
    const int j0  = kt * KTILE;
    const int cur = kt & 1;
    (void)cur;

    // ---- obtain raw K/V slice for this thread ----
    float kv[8], vv8[8];
#if HAVE_RAWBUF
    wait_copies();        // issuing wave(s) drain TENSORcnt/ASYNCcnt
    __syncthreads();      // everyone's copies visible; also fences previous compute
    #pragma unroll
    for (int e = 0; e < 8; ++e) {
      kv[e]  = Kraw[cur][sr * D + sc + e];
      vv8[e] = Vraw[cur][sr * D + sc + e];
    }
    if (kt + 1 < nkt) {   // stream next tile into the other buffer during compute
#if HAVE_TDM
      if (wave == 0) {
        tdm_copy_tile(k + base + (size_t)(j0 + KTILE) * D, &Kraw[cur ^ 1][0]);
        tdm_copy_tile(v + base + (size_t)(j0 + KTILE) * D, &Vraw[cur ^ 1][0]);
      }
#else
      const size_t nbase = base + (size_t)(j0 + KTILE + sr) * D + sc;
      async_copy32B(k + nbase, &Kraw[cur ^ 1][sr * D + sc]);
      async_copy32B(v + nbase, &Vraw[cur ^ 1][sr * D + sc]);
#endif
    }
#else
    __syncthreads();      // fence previous compute before re-staging LDS tiles
    const float* kp = k + base + (size_t)(j0 + sr) * D + sc;
    const float* vp = v + base + (size_t)(j0 + sr) * D + sc;
    #pragma unroll
    for (int e = 0; e < 8; ++e) { kv[e] = kp[e]; vv8[e] = vp[e]; }
    if (kt + 1 < nkt) {
      __builtin_prefetch(kp + KTILE * D, 0, 0);
      __builtin_prefetch(vp + KTILE * D, 0, 0);
    }
#endif

    // ---- normalize K row, stage bf16 K and transposed V ----
    float kss = 0.f;
    #pragma unroll
    for (int e = 0; e < 8; ++e) kss += kv[e] * kv[e];
    kss += __shfl_xor(kss, 1);
    kss += __shfl_xor(kss, 2);
    kss += __shfl_xor(kss, 4);
    const float kinv = 1.0f / fmaxf(sqrtf(kss), EPS);
    #pragma unroll
    for (int e = 0; e < 8; ++e) {
      Klds[sr * D + sc + e]      = (__bf16)(kv[e] * kinv * ksc[e]);
      Vtlds[(sc + e) * VST + sr] = (__bf16)vv8[e];
    }
    if (tid < KTILE) Mlds[tid] = mask[(size_t)bidx * S + j0 + tid];
    __syncthreads();

    // ---- S = (8 * q_hat * q_scale) . (k_hat * k_scale)^T ----
    v8f s0v = {}, s1v = {};
    {
      const v16bf b00 = __builtin_shufflevector(*(const v8bf*)&Klds[(ln)      * D +      hl * 8], *(const v8bf*)&Klds[(ln)      * D + 16 + hl * 8], 0, 1, 2, 3, 4, 5, 6, 7, 8, 9, 10, 11, 12, 13, 14, 15);
      const v16bf b01 = __builtin_shufflevector(*(const v8bf*)&Klds[(ln)      * D + 32 + hl * 8], *(const v8bf*)&Klds[(ln)      * D + 48 + hl * 8], 0, 1, 2, 3, 4, 5, 6, 7, 8, 9, 10, 11, 12, 13, 14, 15);
      const v16bf b10 = __builtin_shufflevector(*(const v8bf*)&Klds[(16 + ln) * D +      hl * 8], *(const v8bf*)&Klds[(16 + ln) * D + 16 + hl * 8], 0, 1, 2, 3, 4, 5, 6, 7, 8, 9, 10, 11, 12, 13, 14, 15);
      const v16bf b11 = __builtin_shufflevector(*(const v8bf*)&Klds[(16 + ln) * D + 32 + hl * 8], *(const v8bf*)&Klds[(16 + ln) * D + 48 + hl * 8], 0, 1, 2, 3, 4, 5, 6, 7, 8, 9, 10, 11, 12, 13, 14, 15);
      s0v = WMMA_BF16(aQ0, b00, s0v);
      s0v = WMMA_BF16(aQ1, b01, s0v);
      s1v = WMMA_BF16(aQ0, b10, s1v);
      s1v = WMMA_BF16(aQ1, b11, s1v);
    }

    // ---- mask + exp(s - SMAX): static shift is exact for bounded cosine-sim logits ----
    __bf16* ps = &Pscr[wave * 16 * KTILE];
    #pragma unroll
    for (int vv = 0; vv < 8; ++vv) {
      const int row = vv + 8 * hl;
      const int qi  = qb + wave * 16 + row;
      float a0 = s0v[vv];
      float a1 = s1v[vv];
      if ((j0 + ln)      > qi || !Mlds[ln])      a0 = NEGMAX;
      if ((j0 + 16 + ln) > qi || !Mlds[16 + ln]) a1 = NEGMAX;
      const float p0 = __expf(a0 - SMAX);
      const float p1 = __expf(a1 - SMAX);
      ps[row * KTILE + ln]      = (__bf16)p0;
      ps[row * KTILE + 16 + ln] = (__bf16)p1;
    }

    // reload P as 16x32 bf16 A-fragment (same-wave DS ops are ordered)
    const v8bf plo = *(const v8bf*)&ps[ln * KTILE + hl * 16];
    const v8bf phi = *(const v8bf*)&ps[ln * KTILE + hl * 16 + 8];
    const v16bf aP = __builtin_shufflevector(plo, phi,
        0, 1, 2, 3, 4, 5, 6, 7, 8, 9, 10, 11, 12, 13, 14, 15);

    // ---- O += P.V ; row-sum accumulated by an extra ones-matrix WMMA ----
    {
      const v16bf bv0 = *(const v16bf*)&Vtlds[(0 * 16 + ln) * VST + hl * 16];
      const v16bf bv1 = *(const v16bf*)&Vtlds[(1 * 16 + ln) * VST + hl * 16];
      const v16bf bv2 = *(const v16bf*)&Vtlds[(2 * 16 + ln) * VST + hl * 16];
      const v16bf bv3 = *(const v16bf*)&Vtlds[(3 * 16 + ln) * VST + hl * 16];
      o0   = WMMA_BF16(aP, bv0, o0);
      o1   = WMMA_BF16(aP, bv1, o1);
      o2   = WMMA_BF16(aP, bv2, o2);
      o3   = WMMA_BF16(aP, bv3, o3);
      osum = WMMA_BF16(aP, vones, osum);
    }
  }

  // ---- epilogue: divide by row-sum (softmax denominator), write fp32 ----
  #pragma unroll
  for (int vv = 0; vv < 8; ++vv) {
    const int qi = qb + wave * 16 + vv + 8 * hl;
    const float invl = 1.0f / osum[vv];
    float* op = out + base + (size_t)qi * D + ln;
    op[0]  = o0[vv] * invl;
    op[16] = o1[vv] * invl;
    op[32] = o2[vv] * invl;
    op[48] = o3[vv] * invl;
  }
}

extern "C" void kernel_launch(void* const* d_in, const int* in_sizes, int n_in,
                              void* d_out, int out_size, void* d_ws, size_t ws_size,
                              hipStream_t stream) {
  (void)in_sizes; (void)n_in; (void)d_ws; (void)ws_size; (void)out_size;
  const float*         q  = (const float*)d_in[0];
  const float*         k  = (const float*)d_in[1];
  const float*         v  = (const float*)d_in[2];
  const float*         qs = (const float*)d_in[3];
  const float*         ks = (const float*)d_in[4];
  const unsigned char* mk = (const unsigned char*)d_in[5];
  float* out = (float*)d_out;

  dim3 grid(S / QTILE, B * H);
  attn_cossim_fwd<<<grid, 256, 0, stream>>>(q, k, v, qs, ks, mk, out);
}
